// UniCrystalFormer_53008486367731
// MI455X (gfx1250) — matrix operand order repaired and checked by reading, not compiled, not run on hardware
//
#include <hip/hip_runtime.h>
#include <cstdint>

#define NN 5000
#define NE 80000
#define NG 64
#define CH 16000   // edge chunk: 5 chunks of 16000 (multiple of 16)

typedef __bf16 bf16_t;
typedef bf16_t bf16x16 __attribute__((ext_vector_type(16)));
typedef float  f32x8   __attribute__((ext_vector_type(8)));
typedef unsigned short u16;

union Frag { bf16x16 v; u16 u[16]; uint4 q[2]; };

__device__ __forceinline__ float b2f(u16 x){ return __uint_as_float(((unsigned int)x) << 16); }
__device__ __forceinline__ u16 f2b_d(float f){
  unsigned int x = __float_as_uint(f);
  x += 0x7FFFu + ((x >> 16) & 1u);
  return (u16)(x >> 16);
}
__device__ __forceinline__ float sig_(float x){ return 1.0f / (1.0f + __expf(-x)); }
__device__ __forceinline__ float silu_(float x){ return x * sig_(x); }

__device__ __forceinline__ float bred128(float v, float* red){
  int c = threadIdx.x;
  red[c] = v; __syncthreads();
  #pragma unroll
  for (int s = 64; s > 0; s >>= 1){
    if (c < s) red[c] += red[c + s];
    __syncthreads();
  }
  float r = red[0];
  __syncthreads();
  return r;
}

// ---------------------------------------------------------------------------
// Weight repack: f32 [K,N] row-major -> bf16 in exact B-fragment order:
// P[((tk*tilesN16 + tn)*32 + lane)*16 + j] = W[tk*32 + (lane>>4)*16 + j, tn*16 + (lane&15)]
// so the GEMM loads each B fragment as two aligned b128s.
// ---------------------------------------------------------------------------
__global__ void repackW_kernel(const float* __restrict__ W, u16* __restrict__ P, int K, int N){
  int i = blockIdx.x * blockDim.x + threadIdx.x;
  if (i >= K * N) return;
  int j   = i & 15;
  int L   = (i >> 4) & 31;
  int t16 = i >> 9;
  int tilesN16 = N >> 4;
  int tk = t16 / tilesN16, tn = t16 % tilesN16;
  int k = tk * 32 + ((L >> 4) << 4) + j;
  int n = tn * 16 + (L & 15);
  P[i] = f2b_d(W[(size_t)k * N + n]);
}

// ---------------------------------------------------------------------------
// bf16 WMMA GEMM: C[M,N] = act(A[M,K] @ W[K,N] + bias)
// one wave -> 16x64 C tile: 1 A-fragment (2 x b128) reused by 4 WMMAs,
// 4 B fragments (2 x b128 each) from the pre-swizzled weight buffer.
// Fragment k order (A and B): contiguous, lane half h holds k 16h..16h+15.
// ---------------------------------------------------------------------------
__global__ __launch_bounds__(256)
void gemm_bf16_kernel(const u16* __restrict__ A, const u16* __restrict__ Wp,
                      const float* __restrict__ bias,
                      float* __restrict__ outF, u16* __restrict__ outB,
                      int M, int K, int N, int act)
{
  const int wave = threadIdx.x >> 5;
  const int lane = threadIdx.x & 31;
  const int tilesN64 = N >> 6;
  const int tilesN16 = N >> 4;
  const int tilesM = (M + 15) >> 4;
  int tile = blockIdx.x * 8 + wave;
  if (tile >= tilesM * tilesN64) return;
  const int tm = tile / tilesN64;
  const int tn = tile % tilesN64;

  int am = tm * 16 + (lane & 15);
  if (am >= M) am = M - 1;
  const int khalfA = (lane >> 4) << 4;           // lanes 16-31 hold K+16..31
  const u16* arow  = A + (size_t)am * K + khalfA;
  const u16* bbase = Wp + (((size_t)tn * 4) << 9) + (lane << 4);
  const size_t bstep = (size_t)tilesN16 << 9;    // one K-step of packed weights

  f32x8 acc[4] = {{}, {}, {}, {}};
  const int ksteps = K >> 5;
  for (int ks = 0; ks < ksteps; ++ks) {
    const u16* ap = arow + (ks << 5);
    Frag af;
    af.q[0] = ((const uint4*)ap)[0];             // K khalf+0..7
    af.q[1] = ((const uint4*)(ap + 8))[0];       // K khalf+8..15
    const u16* bp = bbase + ks * bstep;
    #pragma unroll
    for (int t = 0; t < 4; ++t) {
      Frag bf;
      const uint4* pq = (const uint4*)(bp + (t << 9));
      bf.q[0] = pq[0];
      bf.q[1] = pq[1];
      acc[t] = __builtin_amdgcn_wmma_f32_16x16x32_bf16(false, af.v, false, bf.v, (short)0, acc[t], false, false);
    }
    if (ks + 1 < ksteps) {
      __builtin_prefetch(ap + 32, 0, 3);
      __builtin_prefetch(bp + bstep, 0, 3);
    }
  }

  const int rbase = tm * 16 + ((lane >> 4) << 3);
  const int nb = lane & 15;
  #pragma unroll
  for (int t = 0; t < 4; ++t) {
    int n0 = tn * 64 + t * 16 + nb;
    float bb = bias ? bias[n0] : 0.f;
    #pragma unroll
    for (int v = 0; v < 8; ++v) {
      int r = rbase + v;
      if (r < M) {
        float x = acc[t][v] + bb;
        if (act == 1) x = silu_(x);
        size_t o = (size_t)r * N + n0;
        if (outF) outF[o] = x;
        if (outB) outB[o] = f2b_d(x);
      }
    }
  }
}

// ---------------------------------------------------------------------------
// elementwise / helper kernels
// ---------------------------------------------------------------------------
__global__ void f2b_kernel(const float* __restrict__ s, u16* __restrict__ d, long n){
  long i = (long)blockIdx.x * blockDim.x + threadIdx.x;
  if (i < n) d[i] = f2b_d(s[i]);
}

__global__ void add_kernel(float* a, const float* b, int n){
  int i = blockIdx.x * blockDim.x + threadIdx.x;
  if (i < n) a[i] += b[i];
}

__global__ void count_kernel(const int* batch, float* cnt, int n){
  int i = blockIdx.x * blockDim.x + threadIdx.x;
  if (i < n) atomicAdd(&cnt[batch[i]], 1.0f);
}

__global__ void encode_prep_kernel(const int* __restrict__ an, const float* __restrict__ meg,
                                   const float* __restrict__ etab, const float* __restrict__ wm,
                                   const float* __restrict__ bm,
                                   float* embF, float* megF, u16* catB, int n)
{
  int i = blockIdx.x * blockDim.x + threadIdx.x;
  if (i >= n * 128) return;
  int node = i >> 7, c = i & 127;
  float e = etab[an[node] * 128 + c];
  float m = bm[c];
  const float* mi = meg + node * 16;
  #pragma unroll
  for (int k = 0; k < 16; ++k) m += mi[k] * wm[k * 128 + c];
  embF[i] = e; megF[i] = m;
  catB[(size_t)node * 256 + c]       = f2b_d(e);
  catB[(size_t)node * 256 + 128 + c] = f2b_d(m);
}

__global__ void encode_fuse_kernel(const float* gateF, const float* embF, const float* megF,
                                   u16* outB, int n){
  int i = blockIdx.x * blockDim.x + threadIdx.x;
  if (i >= n * 128) return;
  float g = sig_(gateF[i]);
  outB[i] = f2b_d(g * embF[i] + (1.0f - g) * megF[i]);
}

__global__ __launch_bounds__(128)
void ln128_kernel(const float* __restrict__ x, const float* __restrict__ g,
                  const float* __restrict__ b, u16* outB, float* outF, int M)
{
  __shared__ float red[128];
  int row = blockIdx.x, c = threadIdx.x;
  float v = x[(size_t)row * 128 + c];
  float mu = bred128(v, red) * (1.0f / 128.0f);
  float d = v - mu;
  float var = bred128(d * d, red) * (1.0f / 128.0f);
  float y = g[c] * d * rsqrtf(var + 1e-5f) + b[c];
  if (outB) outB[(size_t)row * 128 + c] = f2b_d(y);
  if (outF) outF[(size_t)row * 128 + c] = y;
}

__global__ void rbf_kernel(const float* __restrict__ evec, u16* out, int E){
  int i = blockIdx.x * blockDim.x + threadIdx.x;
  if (i >= E * 128) return;
  int e = i >> 7, c = i & 127;
  const float* v = evec + (size_t)e * 3;
  float d = sqrtf(v[0]*v[0] + v[1]*v[1] + v[2]*v[2]);
  float t = d - (8.0f / 127.0f) * (float)c;
  out[i] = f2b_d(__expf(-(127.0f / 8.0f) * t * t));
}

__global__ void cat3_kernel(const u16* __restrict__ hB, const u16* __restrict__ eaB,
                            const int* __restrict__ src, const int* __restrict__ dst,
                            u16* out, int E)
{
  long i = (long)blockIdx.x * blockDim.x + threadIdx.x;
  if (i >= (long)E * 384) return;
  int e = (int)(i / 384), j = (int)(i % 384);
  u16 v;
  if (j < 128)      v = hB[(size_t)dst[e] * 128 + j];
  else if (j < 256) v = hB[(size_t)src[e] * 128 + (j - 128)];
  else              v = eaB[(size_t)e * 128 + (j - 256)];
  out[i] = v;
}

__global__ __launch_bounds__(128)
void colstats_kernel(const float* __restrict__ x, float* sum, float* sq, int M, int rpb){
  int c = threadIdx.x;
  int r0 = blockIdx.x * rpb;
  int r1 = r0 + rpb; if (r1 > M) r1 = M;
  float s = 0.f, q = 0.f;
  for (int r = r0; r < r1; ++r){ float v = x[(size_t)r * 128 + c]; s += v; q += v * v; }
  atomicAdd(&sum[c], s); atomicAdd(&sq[c], q);
}

__global__ void cart_scatter_kernel(const float* __restrict__ eij, const float* __restrict__ msg,
                                    const float* __restrict__ sum, const float* __restrict__ sq,
                                    const float* __restrict__ bng, const float* __restrict__ bnb,
                                    const float* __restrict__ cdist, const int* __restrict__ dst,
                                    float* aggr, int E)
{
  int i = blockIdx.x * blockDim.x + threadIdx.x;
  if (i >= E * 128) return;
  int e = i >> 7, c = i & 127;
  float mu = sum[c] / (float)E;
  float var = sq[c] / (float)E - mu * mu;
  float xn = bng[c] * (eij[i] - mu) * rsqrtf(var + 1e-5f) + bnb[c];
  float env = 0.5f * (__cosf(cdist[e] * 0.39269908169872415f) + 1.0f);
  atomicAdd(&aggr[(size_t)dst[e] * 128 + c], sig_(xn) * env * msg[i]);
}

__global__ void cart_finish_kernel(const float* __restrict__ aggr, const float* __restrict__ sum,
                                   const float* __restrict__ sq, const float* __restrict__ g,
                                   const float* __restrict__ b, const float* __restrict__ h,
                                   float* out, int n)
{
  int i = blockIdx.x * blockDim.x + threadIdx.x;
  if (i >= n * 128) return;
  int c = i & 127;
  float mu = sum[c] / (float)n;
  float var = sq[c] / (float)n - mu * mu;
  float xn = g[c] * (aggr[i] - mu) * rsqrtf(var + 1e-5f) + b[c];
  out[i] = silu_(xn) + h[i];
}

__global__ void mcat_kernel(const float* __restrict__ v, const u16* __restrict__ eeB,
                            const int* __restrict__ src, const int* __restrict__ dst,
                            u16* out, int e0, int ec)
{
  long i = (long)blockIdx.x * blockDim.x + threadIdx.x;
  long tot = (long)ec * 4 * 384;
  if (i >= tot) return;
  int j = (int)(i % 384);
  long row = i / 384;
  int hd = (int)(row & 3);
  int e = e0 + (int)(row >> 2);
  u16 o;
  if (j < 128)      o = f2b_d(v[(size_t)dst[e] * 512 + hd * 128 + j]);
  else if (j < 256) o = f2b_d(v[(size_t)src[e] * 512 + hd * 128 + (j - 128)]);
  else              o = eeB[(size_t)e * 512 + hd * 128 + (j - 256)];
  out[i] = o;
}

__global__ __launch_bounds__(128)
void alpha_gate_kernel(const float* __restrict__ q, const float* __restrict__ k,
                       const u16* __restrict__ eeB, const float* __restrict__ ag,
                       const float* __restrict__ ab, const int* __restrict__ src,
                       const int* __restrict__ dst, u16* mupd, int e0)
{
  __shared__ float red[128];
  __shared__ float al[384];
  int row = blockIdx.x;
  int hd = row & 3;
  int e  = e0 + (row >> 2);
  int c  = threadIdx.x;
  int dn = dst[e], sn = src[e];
  float qi = q[(size_t)dn * 512 + hd * 128 + c];
  float ki = k[(size_t)dn * 512 + hd * 128 + c];
  float kj = k[(size_t)sn * 512 + hd * 128 + c];
  float ee = b2f(eeB[(size_t)e * 512 + hd * 128 + c]);
  const float sc = 0.05103103630798287f;  // 1/sqrt(384)
  float a0 = qi * ki * sc, a1 = qi * kj * sc, a2 = qi * ee * sc;
  al[c] = a0; al[128 + c] = a1; al[256 + c] = a2;
  float mu = bred128(a0 + a1 + a2, red) * (1.0f / 384.0f);
  float d0 = a0 - mu, d1 = a1 - mu, d2 = a2 - mu;
  float var = bred128(d0*d0 + d1*d1 + d2*d2, red) * (1.0f / 384.0f);
  float rstd = rsqrtf(var + 1e-5f);
  size_t base = (size_t)row * 384;
  #pragma unroll
  for (int t = 0; t < 3; ++t){
    int j = t * 128 + c;
    float gate = sig_(ag[j] * (al[j] - mu) * rstd + ab[j]);
    mupd[base + j] = f2b_d(b2f(mupd[base + j]) * gate);
  }
}

__global__ __launch_bounds__(128)
void mln_scatter_kernel(const float* __restrict__ mlin, const float* __restrict__ g,
                        const float* __restrict__ b, const int* __restrict__ dst,
                        float* outsum, int e0)
{
  __shared__ float red[128];
  int row = blockIdx.x, c = threadIdx.x;
  float v = mlin[(size_t)row * 128 + c];
  float mu = bred128(v, red) * (1.0f / 128.0f);
  float d = v - mu;
  float var = bred128(d * d, red) * (1.0f / 128.0f);
  float y = g[c] * d * rsqrtf(var + 1e-5f) + b[c];
  int e = e0 + (row >> 2);
  atomicAdd(&outsum[(size_t)dst[e] * 128 + c], 0.25f * y);   // 0.25 = head mean
}

__global__ __launch_bounds__(128)
void beta_kernel(const float* __restrict__ outsum, const float* __restrict__ xr,
                 const float* __restrict__ bw, float* hg, int n)
{
  __shared__ float red[128];
  int node = blockIdx.x, c = threadIdx.x;
  float o = outsum[(size_t)node * 128 + c];
  float x = xr[(size_t)node * 128 + c];
  float s = bred128(o * bw[c] + x * bw[128 + c] + (o - x) * bw[256 + c], red);
  float beta = sig_(s);
  hg[(size_t)node * 128 + c] = beta * x + (1.0f - beta) * o;
}

__global__ void gn_sum_kernel(const float* x, const int* batch, float* gsum, int n){
  int i = blockIdx.x * blockDim.x + threadIdx.x;
  if (i >= n * 128) return;
  atomicAdd(&gsum[(size_t)batch[i >> 7] * 128 + (i & 127)], x[i]);
}
__global__ void gn_dev_kernel(const float* x, const int* batch, const float* gsum,
                              const float* cnt, const float* ms, float* tmp, float* gsq, int n){
  int i = blockIdx.x * blockDim.x + threadIdx.x;
  if (i >= n * 128) return;
  int g = batch[i >> 7], c = i & 127;
  float mean = gsum[g * 128 + c] / fmaxf(cnt[g], 1.0f);
  float d = x[i] - mean * ms[c];
  tmp[i] = d;
  atomicAdd(&gsq[g * 128 + c], d * d);
}
__global__ void gn_fin_kernel(const float* tmp, const int* batch, const float* gsq,
                              const float* cnt, const float* gg, const float* bb,
                              const float* resid, float* out, int n){
  int i = blockIdx.x * blockDim.x + threadIdx.x;
  if (i >= n * 128) return;
  int g = batch[i >> 7], c = i & 127;
  float var = gsq[g * 128 + c] / fmaxf(cnt[g], 1.0f);
  out[i] = resid[i] + gg[c] * tmp[i] * rsqrtf(var + 1e-5f) + bb[c];
}

__global__ __launch_bounds__(256)
void set2set_kernel(const float* __restrict__ x, const int* __restrict__ batch,
                    const float* __restrict__ wih, const float* __restrict__ whh,
                    const float* __restrict__ bih, const float* __restrict__ bhh,
                    float* h, float* c, float* q, float* r, float* ebuf, float* gates)
{
  __shared__ float emax_s[NG];
  __shared__ float esum_s[NG];
  const int tid = threadIdx.x;
  for (int step = 0; step < 3; ++step){
    for (int idx = tid; idx < NG * 512; idx += 256){
      int g = idx >> 9, j = idx & 511;
      float acc = bih[j] + bhh[j];
      const float* qr = q + g * 256;
      const float* wi = wih + (size_t)j * 256;
      for (int kk = 0; kk < 256; ++kk) acc += qr[kk] * wi[kk];
      const float* hr = h + g * 128;
      const float* wh = whh + (size_t)j * 128;
      for (int kk = 0; kk < 128; ++kk) acc += hr[kk] * wh[kk];
      gates[idx] = acc;
    }
    __syncthreads();
    for (int idx = tid; idx < NG * 128; idx += 256){
      int g = idx >> 7, cc = idx & 127;
      const float* gr = gates + g * 512;
      float cn = sig_(gr[128 + cc]) * c[idx] + sig_(gr[cc]) * tanhf(gr[256 + cc]);
      c[idx] = cn;
      h[idx] = sig_(gr[384 + cc]) * tanhf(cn);
    }
    __syncthreads();
    for (int n = tid; n < NN; n += 256){
      const float* xr = x + (size_t)n * 128;
      const float* hr = h + batch[n] * 128;
      float s = 0.f;
      for (int kk = 0; kk < 128; ++kk) s += xr[kk] * hr[kk];
      ebuf[n] = s;
    }
    __syncthreads();
    if (tid < NG){
      float mx = -1e30f;
      for (int n = 0; n < NN; ++n) if (batch[n] == tid) mx = fmaxf(mx, ebuf[n]);
      float sm = 0.f;
      for (int n = 0; n < NN; ++n) if (batch[n] == tid) sm += __expf(ebuf[n] - mx);
      emax_s[tid] = mx; esum_s[tid] = fmaxf(sm, 1e-30f);
    }
    __syncthreads();
    for (int idx = tid; idx < NG * 128; idx += 256) r[idx] = 0.f;
    __syncthreads();
    for (int n = tid; n < NN; n += 256){
      int g = batch[n];
      float a = __expf(ebuf[n] - emax_s[g]) / esum_s[g];
      const float* xr = x + (size_t)n * 128;
      float* rr = r + (size_t)g * 128;
      for (int kk = 0; kk < 128; ++kk) atomicAdd(&rr[kk], a * xr[kk]);
    }
    __syncthreads();
    for (int idx = tid; idx < NG * 256; idx += 256){
      int g = idx >> 8, j = idx & 255;
      q[idx] = (j < 128) ? h[g * 128 + j] : r[g * 128 + (j - 128)];
    }
    __syncthreads();
  }
}

__global__ void fc2_kernel(const float* __restrict__ a, const float* __restrict__ w,
                           const float* __restrict__ b, float* out){
  int g = blockIdx.x * blockDim.x + threadIdx.x;
  if (g >= NG) return;
  float s = b[0];
  for (int k = 0; k < 128; ++k) s += a[(size_t)g * 128 + k] * w[k];
  out[g] = s;
}

// ---------------------------------------------------------------------------
extern "C" void kernel_launch(void* const* d_in, const int* in_sizes, int n_in,
                              void* d_out, int out_size, void* d_ws, size_t ws_size,
                              hipStream_t stream)
{
  (void)in_sizes; (void)n_in; (void)out_size;
  int pi = 0;
  const int*   an    = (const int*)  d_in[pi++];
  const float* megi  = (const float*)d_in[pi++];
  const int*   eidx  = (const int*)  d_in[pi++];
  const float* evec  = (const float*)d_in[pi++];
  const float* cdist = (const float*)d_in[pi++];
  const int*   batch = (const int*)  d_in[pi++];
  const int* srcv = eidx;
  const int* dstv = eidx + NE;
  auto F = [&]() -> const float* { return (const float*)d_in[pi++]; };

  // params in pytree (alphabetical dict-key) order
  const float* atom_emb = F();
  const float *c_a1b[3], *c_a1w[3], *c_a2b[3], *c_a2w[3], *c_bnb[3], *c_bng[3],
              *c_bneb[3], *c_bneg[3], *c_g1b[3], *c_g1w[3], *c_g2b[3], *c_g2w[3];
  for (int i = 0; i < 3; ++i){
    c_a1b[i]=F(); c_a1w[i]=F(); c_a2b[i]=F(); c_a2w[i]=F();
    c_bnb[i]=F(); c_bng[i]=F(); c_bneb[i]=F(); c_bneg[i]=F();
    c_g1b[i]=F(); c_g1w[i]=F(); c_g2b[i]=F(); c_g2w[i]=F();
  }
  const float* enc1b=F(); const float* enc1w=F();
  const float* enc2b=F(); const float* enc2w=F();
  const float* enclnb=F(); const float* enclng=F();
  const float* fc1b=F(); const float* fc1w=F();
  const float* fc2b=F(); const float* fc2w=F();
  const float *f1b[3], *f1w[3], *f2b[3], *f2w[3];
  for (int i = 0; i < 3; ++i){ f1b[i]=F(); f1w[i]=F(); f2b[i]=F(); f2w[i]=F(); }
  const float* gateb=F(); const float* gatew=F();
  const float *gAb[3], *gAg[3], *gAms[3];
  for (int i = 0; i < 3; ++i){ gAb[i]=F(); gAg[i]=F(); gAms[i]=F(); }
  const float *gFb[3], *gFg[3], *gFms[3];
  for (int i = 0; i < 3; ++i){ gFb[i]=F(); gFg[i]=F(); gFms[i]=F(); }
  const float *m_alnb[3],*m_alng[3],*m_bw[3],*m_eb[3],*m_ew[3],*m_kb[3],*m_kw[3],
              *m_mlb[3],*m_mlw[3],*m_mlnb[3],*m_mlng[3],*m_mub[3],*m_muw[3],
              *m_qb[3],*m_qw[3],*m_skb[3],*m_skw[3],*m_vb[3],*m_vw[3];
  for (int i = 0; i < 3; ++i){
    m_alnb[i]=F(); m_alng[i]=F(); m_bw[i]=F();
    m_eb[i]=F(); m_ew[i]=F(); m_kb[i]=F(); m_kw[i]=F();
    m_mlb[i]=F(); m_mlw[i]=F(); m_mlnb[i]=F(); m_mlng[i]=F();
    m_mub[i]=F(); m_muw[i]=F();
    m_qb[i]=F(); m_qw[i]=F(); m_skb[i]=F(); m_skw[i]=F();
    m_vb[i]=F(); m_vw[i]=F();
  }
  const float* megb=F(); const float* megw=F();
  const float* rbf1b=F(); const float* rbf1w=F();
  const float* rbf2b=F(); const float* rbf2w=F();
  const float* s_bhh=F(); const float* s_bih=F(); const float* s_whh=F(); const float* s_wih=F();

  // ---- workspace bump allocator ----
  char* wp = (char*)d_ws;
  size_t used = 0;
  auto alloc = [&](size_t bytes) -> char* {
    char* p = wp + used;
    used += (bytes + 255) & ~(size_t)255;
    return p;
  };
  auto aF = [&](size_t n){ return (float*)alloc(n * 4); };
  auto aB = [&](size_t n){ return (u16*)  alloc(n * 2); };

  // bf16 weights (pre-swizzled into fragment order)
  u16* wb_gate = aB(256*128); u16* wb_e1 = aB(128*128); u16* wb_e2 = aB(128*128);
  u16* wb_r1 = aB(128*128);  u16* wb_r2 = aB(128*128); u16* wb_fc1 = aB(256*128);
  u16 *wb_g1[3], *wb_g2[3], *wb_a1[3], *wb_a2[3], *wb_q[3], *wb_k[3], *wb_v[3],
      *wb_ee[3], *wb_up[3], *wb_ml[3], *wb_sk[3], *wb_f1[3], *wb_f2[3];
  for (int i = 0; i < 3; ++i){
    wb_g1[i]=aB(384*128); wb_g2[i]=aB(128*128); wb_a1[i]=aB(384*128); wb_a2[i]=aB(128*128);
    wb_q[i]=aB(128*512); wb_k[i]=aB(128*512); wb_v[i]=aB(128*512); wb_ee[i]=aB(128*512);
    wb_up[i]=aB(384*384); wb_ml[i]=aB(384*128); wb_sk[i]=aB(128*128);
    wb_f1[i]=aB(128*256); wb_f2[i]=aB(256*128);
  }
  // node buffers
  float* h_f = aF((size_t)NN*128); u16* h_b = aB((size_t)NN*128);
  float* n128a = aF((size_t)NN*128); float* n128b = aF((size_t)NN*128);
  float* n128c = aF((size_t)NN*128); float* n128d = aF((size_t)NN*128);
  u16* n128bf = aB((size_t)NN*128); u16* n256bf = aB((size_t)NN*256);
  float* qbuf = aF((size_t)NN*512); float* kbuf = aF((size_t)NN*512); float* vbuf = aF((size_t)NN*512);
  // edge buffers
  u16* ea_bf  = aB((size_t)NE*128);
  u16* rbf_bf = aB((size_t)NE*128);
  u16* cat_bf = aB((size_t)NE*384);
  u16* eb128  = aB((size_t)NE*128);
  float* ef_a = aF((size_t)NE*128);
  float* ef_b = aF((size_t)NE*128);
  u16* ee_bf  = aB((size_t)NE*512);
  u16* mcat_bf = aB((size_t)CH*4*384);
  u16* mupd_bf = aB((size_t)CH*4*384);
  float* mlin_f = aF((size_t)CH*4*128);
  // stats / small
  float* csum = aF(128); float* csq = aF(128);
  float* gnsum = aF((size_t)NG*128); float* gnsq = aF((size_t)NG*128);
  float* cnt = aF(NG);
  float* s_h = aF((size_t)NG*128); float* s_c = aF((size_t)NG*128);
  float* s_q = aF((size_t)NG*256); float* s_r = aF((size_t)NG*128);
  float* s_e = aF(NN); float* s_g = aF((size_t)NG*512);
  u16* q_bf = aB((size_t)NG*256); float* fc1o = aF((size_t)NG*128);

  if (used > ws_size) return;

  auto conv = [&](const float* s, u16* d, long n){
    f2b_kernel<<<dim3((unsigned)((n + 255) / 256)), dim3(256), 0, stream>>>(s, d, n);
  };
  auto repack = [&](const float* w, u16* d, int K, int N){
    repackW_kernel<<<dim3((K * N + 255) / 256), dim3(256), 0, stream>>>(w, d, K, N);
  };
  auto gemm = [&](const u16* A, const u16* W, const float* bias, float* oF, u16* oB,
                  int M, int K, int N, int act){
    int tiles = ((M + 15) / 16) * (N / 64);
    gemm_bf16_kernel<<<dim3((tiles + 7) / 8), dim3(256), 0, stream>>>(A, W, bias, oF, oB, M, K, N, act);
  };
  auto mz = [&](void* p, size_t bytes){ hipMemsetAsync(p, 0, bytes, stream); };
  auto GR = [&](long n){ return dim3((unsigned)((n + 255) / 256)); };

  // ---- weight repack (fragment-order bf16) ----
  repack(gatew, wb_gate, 256, 128); repack(enc1w, wb_e1, 128, 128); repack(enc2w, wb_e2, 128, 128);
  repack(rbf1w, wb_r1, 128, 128);   repack(rbf2w, wb_r2, 128, 128); repack(fc1w, wb_fc1, 256, 128);
  for (int i = 0; i < 3; ++i){
    repack(c_g1w[i], wb_g1[i], 384, 128); repack(c_g2w[i], wb_g2[i], 128, 128);
    repack(c_a1w[i], wb_a1[i], 384, 128); repack(c_a2w[i], wb_a2[i], 128, 128);
    repack(m_qw[i], wb_q[i], 128, 512);   repack(m_kw[i], wb_k[i], 128, 512);
    repack(m_vw[i], wb_v[i], 128, 512);   repack(m_ew[i], wb_ee[i], 128, 512);
    repack(m_muw[i], wb_up[i], 384, 384); repack(m_mlw[i], wb_ml[i], 384, 128);
    repack(m_skw[i], wb_sk[i], 128, 128);
    repack(f1w[i], wb_f1[i], 128, 256);   repack(f2w[i], wb_f2[i], 256, 128);
  }

  // ---- per-graph counts ----
  mz(cnt, NG * 4);
  count_kernel<<<GR(NN), 256, 0, stream>>>(batch, cnt, NN);

  // ---- atom encoder ----
  encode_prep_kernel<<<GR((long)NN*128), 256, 0, stream>>>(an, megi, atom_emb, megw, megb,
                                                           n128a, n128b, n256bf, NN);
  gemm(n256bf, wb_gate, gateb, n128c, nullptr, NN, 256, 128, 0);
  encode_fuse_kernel<<<GR((long)NN*128), 256, 0, stream>>>(n128c, n128a, n128b, n128bf, NN);
  gemm(n128bf, wb_e1, enc1b, n128a, nullptr, NN, 128, 128, 1);
  ln128_kernel<<<dim3(NN), dim3(128), 0, stream>>>(n128a, enclng, enclnb, n128bf, nullptr, NN);
  gemm(n128bf, wb_e2, enc2b, h_f, h_b, NN, 128, 128, 0);

  // ---- edge attr: RBF + MLP ----
  rbf_kernel<<<GR((long)NE*128), 256, 0, stream>>>(evec, rbf_bf, NE);
  gemm(rbf_bf, wb_r1, rbf1b, nullptr, eb128, NE, 128, 128, 1);
  gemm(eb128, wb_r2, rbf2b, nullptr, ea_bf, NE, 128, 128, 0);

  // ---- layers ----
  for (int L = 0; L < 3; ++L){
    conv(h_f, h_b, (long)NN * 128);
    // CartNet layer
    cat3_kernel<<<GR((long)NE*384), 256, 0, stream>>>(h_b, ea_bf, srcv, dstv, cat_bf, NE);
    gemm(cat_bf, wb_g1[L], c_g1b[L], nullptr, eb128, NE, 384, 128, 1);
    gemm(eb128, wb_g2[L], c_g2b[L], ef_a, nullptr, NE, 128, 128, 0);
    gemm(cat_bf, wb_a1[L], c_a1b[L], nullptr, eb128, NE, 384, 128, 1);
    gemm(eb128, wb_a2[L], c_a2b[L], ef_b, nullptr, NE, 128, 128, 0);
    mz(csum, 512); mz(csq, 512);
    colstats_kernel<<<dim3((NE + 511) / 512), dim3(128), 0, stream>>>(ef_a, csum, csq, NE, 512);
    mz(n128a, (size_t)NN * 128 * 4);
    cart_scatter_kernel<<<GR((long)NE*128), 256, 0, stream>>>(ef_a, ef_b, csum, csq,
        c_bneg[L], c_bneb[L], cdist, dstv, n128a, NE);
    mz(csum, 512); mz(csq, 512);
    colstats_kernel<<<dim3((NN + 511) / 512), dim3(128), 0, stream>>>(n128a, csum, csq, NN, 512);
    cart_finish_kernel<<<GR((long)NN*128), 256, 0, stream>>>(n128a, csum, csq,
        c_bng[L], c_bnb[L], h_f, n128a, NN);           // n128a = h_local

    // Matformer layer
    gemm(h_b, wb_q[L], m_qb[L], qbuf, nullptr, NN, 128, 512, 0);
    gemm(h_b, wb_k[L], m_kb[L], kbuf, nullptr, NN, 128, 512, 0);
    gemm(h_b, wb_v[L], m_vb[L], vbuf, nullptr, NN, 128, 512, 0);
    gemm(ea_bf, wb_ee[L], m_eb[L], nullptr, ee_bf, NE, 128, 512, 0);
    mz(n128b, (size_t)NN * 128 * 4);                   // n128b = head-mean segment sum
    for (int c0 = 0; c0 < NE; c0 += CH){
      int ec = (NE - c0 < CH) ? (NE - c0) : CH;
      mcat_kernel<<<GR((long)ec*4*384), 256, 0, stream>>>(vbuf, ee_bf, srcv, dstv, mcat_bf, c0, ec);
      gemm(mcat_bf, wb_up[L], m_mub[L], nullptr, mupd_bf, ec * 4, 384, 384, 0);
      alpha_gate_kernel<<<dim3(ec * 4), dim3(128), 0, stream>>>(qbuf, kbuf, ee_bf,
          m_alng[L], m_alnb[L], srcv, dstv, mupd_bf, c0);
      gemm(mupd_bf, wb_ml[L], m_mlb[L], mlin_f, nullptr, ec * 4, 384, 128, 0);
      mln_scatter_kernel<<<dim3(ec * 4), dim3(128), 0, stream>>>(mlin_f, m_mlng[L], m_mlnb[L],
          dstv, n128b, c0);
    }
    gemm(h_b, wb_sk[L], m_skb[L], n128c, nullptr, NN, 128, 128, 0);   // x_r
    beta_kernel<<<dim3(NN), dim3(128), 0, stream>>>(n128b, n128c, m_bw[L], n128d, NN);

    // combine + GraphNorm (attn)
    add_kernel<<<GR((long)NN*128), 256, 0, stream>>>(n128a, n128d, NN * 128);  // h_comb
    mz(gnsum, (size_t)NG*128*4); mz(gnsq, (size_t)NG*128*4);
    gn_sum_kernel<<<GR((long)NN*128), 256, 0, stream>>>(n128a, batch, gnsum, NN);
    gn_dev_kernel<<<GR((long)NN*128), 256, 0, stream>>>(n128a, batch, gnsum, cnt, gAms[L], n128d, gnsq, NN);
    gn_fin_kernel<<<GR((long)NN*128), 256, 0, stream>>>(n128d, batch, gnsq, cnt, gAg[L], gAb[L], h_f, n128c, NN); // h_attn

    // FFN + GraphNorm (ffn)
    conv(n128c, n128bf, (long)NN * 128);
    gemm(n128bf, wb_f1[L], f1b[L], nullptr, n256bf, NN, 128, 256, 1);
    gemm(n256bf, wb_f2[L], f2b[L], n128d, nullptr, NN, 256, 128, 0);
    mz(gnsum, (size_t)NG*128*4); mz(gnsq, (size_t)NG*128*4);
    gn_sum_kernel<<<GR((long)NN*128), 256, 0, stream>>>(n128d, batch, gnsum, NN);
    gn_dev_kernel<<<GR((long)NN*128), 256, 0, stream>>>(n128d, batch, gnsum, cnt, gFms[L], n128a, gnsq, NN);
    gn_fin_kernel<<<GR((long)NN*128), 256, 0, stream>>>(n128a, batch, gnsq, cnt, gFg[L], gFb[L], n128c, h_f, NN);
  }

  // ---- set2set readout ----
  mz(s_h, (size_t)NG*128*4); mz(s_c, (size_t)NG*128*4); mz(s_q, (size_t)NG*256*4);
  set2set_kernel<<<dim3(1), dim3(256), 0, stream>>>(h_f, batch, s_wih, s_whh, s_bih, s_bhh,
                                                    s_h, s_c, s_q, s_r, s_e, s_g);
  conv(s_q, q_bf, (long)NG * 256);
  gemm(q_bf, wb_fc1, fc1b, fc1o, nullptr, NG, 256, 128, 1);
  fc2_kernel<<<dim3(1), dim3(NG), 0, stream>>>(fc1o, fc2w, fc2b, (float*)d_out);
}
